// Prompt_31404800868863
// MI455X (gfx1250) — matrix operand order repaired and checked by reading, not compiled, not run on hardware
//
#include <hip/hip_runtime.h>
#include <hip/hip_fp16.h>

typedef __attribute__((ext_vector_type(16))) _Float16 v16h;
typedef __attribute__((ext_vector_type(8)))  float    v8f;

#define B_    32
#define N_    4096
#define M_    4096
#define D_    256
#define POOL_ 30
#define LEN_  64
#define TOPK_ 5
#define HIST_ 100
#define L_    (TOPK_ * LEN_)   // 320

// ---- workspace layout (bytes) ----
#define WS_XM    0                        // 32*256 f32   = 32768
#define WS_LOGIT (WS_XM + 32768)          // 960 f32      = 3840
#define WS_SIM   (WS_LOGIT + 3840)        // 960 f32      = 3840
#define WS_SEL   (WS_SIM + 3840)          // 5 int, pad 64
#define WS_KN    (WS_SEL + 64)            // 320*256 f16  = 163840
#define WS_BPT   (WS_KN + 163840)         // 256*320 f16  = 163840

// ---- LDS layout for k_attn (bytes) ----
#define LDS_QNF   0                       // frag-major q: 4*8*32*16 f16 = 32768
#define LDS_SC    32768                   // scores 64*324 f32 = 82944
#define LDS_CHK   (LDS_SC + 82944)        // 32768 (K or V^T chunk)
#define LDS_PF    (LDS_CHK + 32768)       // frag-major probs: 4*10*32*16 f16 = 40960
#define LDS_TOTAL (LDS_PF + 40960)        // 189440

// ---------------- helpers ----------------
__device__ __forceinline__ float block_reduce(float v, float* red) {
    int t = threadIdx.x;
    red[t] = v; __syncthreads();
    for (int s = 128; s > 0; s >>= 1) {
        if (t < s) red[t] += red[t + s];
        __syncthreads();
    }
    float r = red[0]; __syncthreads();
    return r;
}

// ---------------- kernel 1: x_mean ----------------
__global__ void k_mean(const float* __restrict__ x, float* __restrict__ xm) {
    int b = blockIdx.x, d = threadIdx.x;
    const float* p = x + (size_t)b * N_ * D_ + d;
    float s = 0.f;
    for (int n = 0; n < N_; ++n) s += p[(size_t)n * D_];
    xm[b * D_ + d] = s * (1.0f / N_);
}

// ---------------- kernel 2: per-(b,p) MLP relevance + cosine similarity ----------------
__global__ void k_mlp(const float* __restrict__ xm,  const float* __restrict__ pk,
                      const float* __restrict__ W1,  const float* __restrict__ b1,
                      const float* __restrict__ lng, const float* __restrict__ lnb,
                      const float* __restrict__ W2,  const float* __restrict__ b2,
                      float* __restrict__ logits,    float* __restrict__ sim) {
    __shared__ float sfeat[D_];
    __shared__ float red[256];
    int bp = blockIdx.x;
    int b = bp / POOL_, p = bp % POOL_;
    int j = threadIdx.x;
    float xv = xm[b * D_ + j];
    float kv = pk[p * D_ + j];
    sfeat[j] = xv * kv;
    __syncthreads();
    float h = b1[j];
    for (int d = 0; d < D_; ++d) h += sfeat[d] * W1[d * D_ + j];
    float mu  = block_reduce(h, red) * (1.f / D_);
    float var = block_reduce((h - mu) * (h - mu), red) * (1.f / D_);
    float hn  = (h - mu) * rsqrtf(var + 1e-5f) * lng[j] + lnb[j];
    float g   = 0.5f * hn * (1.f + erff(hn * 0.70710678118654752f));
    float logit = block_reduce(g * W2[j], red);
    float dot = block_reduce(xv * kv, red);
    float nx  = block_reduce(xv * xv, red);
    float nk  = block_reduce(kv * kv, red);
    if (j == 0) {
        logits[bp] = logit + b2[0];
        sim[bp] = dot / (fmaxf(sqrtf(nx), 1e-12f) * fmaxf(sqrtf(nk), 1e-12f));
    }
}

// ---------------- kernel 3: diversity, softmax, final scores, top-k ----------------
__global__ void k_select(const float* __restrict__ logits, const float* __restrict__ sim,
                         const float* __restrict__ hist, float* __restrict__ out,
                         int* __restrict__ sel_ws) {
    __shared__ float sdiv[POOL_];
    __shared__ float sfinal[B_][POOL_];
    __shared__ float sbs[POOL_];
    __shared__ int   ssel[TOPK_];
    int t = threadIdx.x;
    if (t < POOL_) {
        float s = 0.f;
        for (int h = 0; h < HIST_; ++h) s += hist[h * POOL_ + t];
        sdiv[t] = 1.f - s * (1.f / HIST_);
    }
    __syncthreads();
    if (t < B_) {
        float mx = -1e30f;
        for (int p = 0; p < POOL_; ++p) mx = fmaxf(mx, logits[t * POOL_ + p]);
        float e[POOL_]; float se = 0.f;
        for (int p = 0; p < POOL_; ++p) { e[p] = __expf(logits[t * POOL_ + p] - mx); se += e[p]; }
        float inv = 1.f / se;
        for (int p = 0; p < POOL_; ++p)
            sfinal[t][p] = 0.5f * sim[t * POOL_ + p] + 0.3f * sdiv[p] + 0.2f * e[p] * inv;
    }
    __syncthreads();
    if (t < POOL_) {
        float s = 0.f;
        for (int b = 0; b < B_; ++b) s += sfinal[b][t];
        sbs[t] = s * (1.f / B_);
    }
    __syncthreads();
    if (t == 0) {
        bool used[POOL_];
        for (int p = 0; p < POOL_; ++p) used[p] = false;
        for (int k = 0; k < TOPK_; ++k) {
            int best = 0; float bv = -1e30f;
            for (int p = 0; p < POOL_; ++p)
                if (!used[p] && sbs[p] > bv) { bv = sbs[p]; best = p; }
            used[best] = true; ssel[k] = best; sel_ws[k] = best;
        }
    }
    __syncthreads();
    const size_t OFF_FINAL = (size_t)B_ * M_ * D_;
    const size_t OFF_SEL   = OFF_FINAL + B_ * POOL_;
    const size_t OFF_DIV   = OFF_SEL + B_ * TOPK_;
    for (int i = t; i < B_ * POOL_; i += blockDim.x)
        out[OFF_FINAL + i] = sfinal[i / POOL_][i % POOL_];
    if (t < B_ * TOPK_) out[OFF_SEL + t] = (float)ssel[t % TOPK_];
    if (t < POOL_)      out[OFF_DIV + t] = sdiv[t];
}

// ---------------- kernel 4: gather selected prompts, build f16 K (normalized) and V^T ----------------
__global__ void k_prep(const float* __restrict__ prompt, const int* __restrict__ sel,
                       _Float16* __restrict__ kn, _Float16* __restrict__ bpT) {
    __shared__ float red[256];
    int l = blockIdx.x, d = threadIdx.x;
    int p = sel[l / LEN_];
    float v = prompt[((size_t)p * LEN_ + (l % LEN_)) * D_ + d];
    float ss = block_reduce(v * v, red);
    float scl = 1.f / fmaxf(sqrtf(ss), 1e-12f);
    kn[(size_t)l * D_ + d]  = (_Float16)(v * scl);
    bpT[(size_t)d * L_ + l] = (_Float16)v;
}

// async global->LDS 16B copy (GVS mode: lds addr VGPR, 32-bit offset VGPR, SGPR64 base)
__device__ __forceinline__ void async_b128(unsigned lds, unsigned off, unsigned long long base) {
    asm volatile("global_load_async_to_lds_b128 %0, %1, %2"
                 :: "v"(lds), "v"(off), "s"(base) : "memory");
}
__device__ __forceinline__ void wait_async0() {
    asm volatile("s_wait_asynccnt 0" ::: "memory");
}

// ---------------- kernel 5: depth attention, WMMA f16 ----------------
// block = (batch b, 64 rows of M), 256 threads = 8 waves
__global__ __launch_bounds__(256)
void k_attn(const float* __restrict__ df, const _Float16* __restrict__ knWs,
            const _Float16* __restrict__ bpT, float* __restrict__ out) {
    extern __shared__ char smem[];
    _Float16* qnf = (_Float16*)(smem + LDS_QNF);   // [4 mt][8 kc][32 lanes][16 halfs]
    const int SROW = 324;
    float*    sc  = (float*)(smem + LDS_SC);       // [64][324]
    _Float16* chk = (_Float16*)(smem + LDS_CHK);   // staged K/V^T chunk
    _Float16* pf  = (_Float16*)(smem + LDS_PF);    // [4 mt][10 kc][32 lanes][16 halfs]

    int tid = threadIdx.x;
    int wave = tid >> 5, lane = tid & 31;
    int b = blockIdx.x >> 6;
    int Mbase = (blockIdx.x & 63) * 64;

    unsigned chkBase = (unsigned)(size_t)(void*)chk;   // LDS byte offset (flat addr low 32)

    // fragment slot for this lane's K position (kpos == lane for d = lane + i*32)
    int g  = (lane >> 3) & 1;
    int hi = lane >> 4;
    int kk = lane - g * 8 - hi * 16;
    int hh = lane & 15;  // half index 0..15
    int lgrp = hi * 16;                              // fragment lane group base
    int mt = wave >> 1;                             // this wave's m-tile (phases 0/A/B/C)

    // Phase 0: load 64 q rows, L2-normalize, scatter f16 into fragment-major LDS
    for (int r = 0; r < 8; ++r) {
        int row = wave * 8 + r;
        const float* src = df + ((size_t)b * M_ + Mbase + row) * D_;
        float v[8]; float ss = 0.f;
        #pragma unroll
        for (int i = 0; i < 8; ++i) { v[i] = src[lane + i * 32]; ss += v[i] * v[i]; }
        for (int o = 16; o > 0; o >>= 1) ss += __shfl_xor(ss, o, 32);
        float scl = 1.f / fmaxf(sqrtf(ss), 1e-12f);
        int m = row & 15;
        #pragma unroll
        for (int i = 0; i < 8; ++i)
            qnf[((mt * 8 + i) * 32 + m + lgrp) * 16 + hh] = (_Float16)(v[i] * scl);
    }
    __syncthreads();

    // Phase A: scores = qn @ kn^T * (1/16), 5 L-chunks of 64, async-staged keys
    {
        int lt0 = (wave & 1) * 2;
        for (int lcb = 0; lcb < 5; ++lcb) {
            #pragma unroll
            for (int i = 0; i < 8; ++i)   // 32 KB chunk, 16 B per lane per op
                async_b128(chkBase + tid * 16 + i * 4096,
                           (unsigned)(lcb * 32768 + tid * 16 + i * 4096),
                           (unsigned long long)(size_t)knWs);
            wait_async0();
            __syncthreads();
            v8f acc0 = {0,0,0,0,0,0,0,0}, acc1 = {0,0,0,0,0,0,0,0};
            int n  = lane & 15;
            int kq = (lane >> 4) * 16;
            for (int kc = 0; kc < 8; ++kc) {
                v16h a = *(const v16h*)&qnf[((mt * 8 + kc) * 32 + lane) * 16];
                v16h b0 = *(const v16h*)&chk[((lt0 + 0) * 16 + n) * D_ + kc * 32 + kq];
                acc0 = __builtin_amdgcn_wmma_f32_16x16x32_f16(false, a, false, b0,
                                                              (short)0, acc0, false, false);
                v16h b1v = *(const v16h*)&chk[((lt0 + 1) * 16 + n) * D_ + kc * 32 + kq];
                acc1 = __builtin_amdgcn_wmma_f32_16x16x32_f16(false, a, false, b1v,
                                                              (short)0, acc1, false, false);
            }
            int mo = (lane >> 4) * 8;
            #pragma unroll
            for (int r = 0; r < 8; ++r) {
                sc[(mt * 16 + mo + r) * SROW + lcb * 64 + (lt0 + 0) * 16 + n] = acc0[r] * 0.0625f;
                sc[(mt * 16 + mo + r) * SROW + lcb * 64 + (lt0 + 1) * 16 + n] = acc1[r] * 0.0625f;
            }
            __syncthreads();
        }
    }

    // Phase B: row softmax over L=320, write probs as f16 into fragment-major pf
    for (int r = 0; r < 8; ++r) {
        int row = wave * 8 + r;
        float v[10]; float mx = -1e30f;
        #pragma unroll
        for (int i = 0; i < 10; ++i) { v[i] = sc[row * SROW + lane + i * 32]; mx = fmaxf(mx, v[i]); }
        for (int o = 16; o > 0; o >>= 1) mx = fmaxf(mx, __shfl_xor(mx, o, 32));
        float s = 0.f;
        #pragma unroll
        for (int i = 0; i < 10; ++i) { v[i] = __expf(v[i] - mx); s += v[i]; }
        for (int o = 16; o > 0; o >>= 1) s += __shfl_xor(s, o, 32);
        float inv = 1.f / s;
        int m = row & 15;
        #pragma unroll
        for (int i = 0; i < 10; ++i)
            pf[((mt * 10 + i) * 32 + m + lgrp) * 16 + hh] = (_Float16)(v[i] * inv);
    }
    __syncthreads();

    // Phase C: out = P @ bp, 8 accumulator tiles per wave, async-staged V^T chunks
    {
        v8f acc[8];
        v8f z = {0,0,0,0,0,0,0,0};
        #pragma unroll
        for (int t = 0; t < 8; ++t) acc[t] = z;
        int ncb = (wave & 1) * 128;
        int n   = lane & 15;
        int kq  = (lane >> 4) * 16;
        for (int lcb = 0; lcb < 5; ++lcb) {
            #pragma unroll
            for (int i = 0; i < 8; ++i)   // row d=tid: 128 B from bpT row into chk[tid*64..]
                async_b128(chkBase + tid * 128 + i * 16,
                           (unsigned)(tid * (L_ * 2) + lcb * 128 + i * 16),
                           (unsigned long long)(size_t)bpT);
            wait_async0();
            __syncthreads();
            for (int kc = 0; kc < 2; ++kc) {
                v16h a = *(const v16h*)&pf[((mt * 10 + lcb * 2 + kc) * 32 + lane) * 16];
                #pragma unroll
                for (int nt = 0; nt < 8; ++nt) {
                    v16h bv = *(const v16h*)&chk[(ncb + nt * 16 + n) * 64 + kc * 32 + kq];
                    acc[nt] = __builtin_amdgcn_wmma_f32_16x16x32_f16(false, a, false, bv,
                                                                     (short)0, acc[nt], false, false);
                }
            }
            __syncthreads();
        }
        int mo = (lane >> 4) * 8;
        for (int nt = 0; nt < 8; ++nt) {
            #pragma unroll
            for (int r = 0; r < 8; ++r) {
                out[((size_t)b * M_ + Mbase + mt * 16 + mo + r) * D_ + ncb + nt * 16 + n] =
                    acc[nt][r];
            }
        }
    }
}

// ---------------- host launch ----------------
extern "C" void kernel_launch(void* const* d_in, const int* in_sizes, int n_in,
                              void* d_out, int out_size, void* d_ws, size_t ws_size,
                              hipStream_t stream) {
    (void)in_sizes; (void)n_in; (void)out_size; (void)ws_size;
    const float* x_embed = (const float*)d_in[0];
    const float* depth   = (const float*)d_in[1];
    const float* prompt  = (const float*)d_in[2];
    const float* pkey    = (const float*)d_in[3];
    const float* W1      = (const float*)d_in[4];
    const float* b1      = (const float*)d_in[5];
    const float* lng     = (const float*)d_in[6];
    const float* lnb     = (const float*)d_in[7];
    const float* W2      = (const float*)d_in[8];
    const float* b2      = (const float*)d_in[9];
    const float* hist    = (const float*)d_in[10];
    float* out = (float*)d_out;
    char* ws = (char*)d_ws;
    float*    xm     = (float*)(ws + WS_XM);
    float*    logits = (float*)(ws + WS_LOGIT);
    float*    sim    = (float*)(ws + WS_SIM);
    int*      sel    = (int*)(ws + WS_SEL);
    _Float16* kn     = (_Float16*)(ws + WS_KN);
    _Float16* bpT    = (_Float16*)(ws + WS_BPT);

    size_t shmem = LDS_TOTAL;   // 189440 B dynamic LDS
    (void)hipFuncSetAttribute((const void*)k_attn,
                              hipFuncAttributeMaxDynamicSharedMemorySize, (int)shmem);

    k_mean  <<<B_,             D_,  0,     stream>>>(x_embed, xm);
    k_mlp   <<<B_ * POOL_,     256, 0,     stream>>>(xm, pkey, W1, b1, lng, lnb, W2, b2, logits, sim);
    k_select<<<1,              256, 0,     stream>>>(logits, sim, hist, out, sel);
    k_prep  <<<L_,             D_,  0,     stream>>>(prompt, sel, kn, bpT);
    k_attn  <<<B_ * (M_ / 64), 256, shmem, stream>>>(depth, kn, bpT, out);
}
